// StrataCausalAttention_17179869588
// MI455X (gfx1250) — matrix operand order repaired and checked by reading, not compiled, not run on hardware
//
#include <hip/hip_runtime.h>
#include <stdint.h>

typedef __bf16 bf16_t;
typedef __attribute__((ext_vector_type(16))) __bf16 v16bf;
typedef __attribute__((ext_vector_type(8)))  float  v8f;

// Problem constants
#define BB   2
#define TT   2048
#define CC   768
#define HH   12
#define HD   64
#define MM   (BB*TT)      // 4096
#define NQKV (3*CC)       // 2304

union AF {
    v16bf    v;
    uint32_t u[8];
    bf16_t   e[16];
};

// A-matrix per-lane K pattern (16-bit A 16x32): vgpr v, half h -> k0
__device__ __forceinline__ int apat(int v, int h) {
    return 2 * v + 16 * h;
}
// B-matrix per-lane K pattern (16-bit B 32x16): vgpr v, half h -> k0
__device__ __forceinline__ int bpat(int v, int h) {
    return 2 * v + 16 * h;
}

// ---------------------------------------------------------------------------
// 1) fp32 -> bf16 convert
// ---------------------------------------------------------------------------
__global__ void k_cvt_bf16(const float* __restrict__ src, bf16_t* __restrict__ dst, int n) {
    int i = blockIdx.x * blockDim.x + threadIdx.x;
    if (i < n) dst[i] = (bf16_t)src[i];
}

// ---------------------------------------------------------------------------
// 2) Pack a row-major fp32 weight [K,N] into per-lane B-fragment bf16 layout:
//    out[(nt*KT + kt)*512 + lane*16 + e], e = 2v+odd, k = kt*32 + 2v + 16*half + odd
// ---------------------------------------------------------------------------
__global__ void k_pack_w(const float* __restrict__ W, bf16_t* __restrict__ out, int K, int N) {
    int tid = blockIdx.x * blockDim.x + threadIdx.x;
    int total = K * N;
    if (tid >= total) return;
    int frag = tid >> 9;
    int rem  = tid & 511;
    int lane = rem >> 4;
    int e    = rem & 15;
    int KT   = K >> 5;
    int nt   = frag / KT;
    int kt   = frag - nt * KT;
    int n    = nt * 16 + (lane & 15);
    int h    = lane >> 4;
    int k    = kt * 32 + 2 * (e >> 1) + 16 * h + (e & 1);
    out[tid] = (bf16_t)W[(size_t)k * N + n];
}

// ---------------------------------------------------------------------------
// 3) QKV GEMM, 2x2 register-blocked: one wave computes a 32x32 output tile
//    (4 WMMAs per k-step over 4 fragment loads). Epilogue adds bias and
//    scatters Q,K row-major [B,H,T,64] and V transposed [B,H,64,T].
// ---------------------------------------------------------------------------
__global__ void k_gemm_qkv(const bf16_t* __restrict__ xb,
                           const bf16_t* __restrict__ Wp,
                           const float*  __restrict__ bias,
                           bf16_t* __restrict__ Qb,
                           bf16_t* __restrict__ Kb,
                           bf16_t* __restrict__ Vt) {
    const int KT = CC / 32;                   // 24
    int nt0 = blockIdx.x * 2, mt0 = blockIdx.y * 2;
    int L  = threadIdx.x;
    int lm = L & 15, h = L >> 4;

    v8f acc[2][2];
    acc[0][0] = v8f{}; acc[0][1] = v8f{}; acc[1][0] = v8f{}; acc[1][1] = v8f{};

    const bf16_t* ap0 = xb + (size_t)(mt0 * 16 + lm) * CC;
    const bf16_t* ap1 = ap0 + (size_t)16 * CC;
    const bf16_t* bp0 = Wp + (size_t)(nt0 * KT) * 512 + L * 16;
    const bf16_t* bp1 = bp0 + (size_t)KT * 512;

    for (int kt = 0; kt < KT; ++kt) {
        AF a0, a1, b0, b1;
#pragma unroll
        for (int v = 0; v < 8; ++v) {
            int ka = kt * 32 + apat(v, h);
            a0.u[v] = *(const uint32_t*)(ap0 + ka);
            a1.u[v] = *(const uint32_t*)(ap1 + ka);
            b0.u[v] = *(const uint32_t*)(bp0 + kt * 512 + 2 * v);
            b1.u[v] = *(const uint32_t*)(bp1 + kt * 512 + 2 * v);
        }
        acc[0][0] = __builtin_amdgcn_wmma_f32_16x16x32_bf16(false, a0.v, false, b0.v,
                                                            (short)0, acc[0][0], false, false);
        acc[0][1] = __builtin_amdgcn_wmma_f32_16x16x32_bf16(false, a0.v, false, b1.v,
                                                            (short)0, acc[0][1], false, false);
        acc[1][0] = __builtin_amdgcn_wmma_f32_16x16x32_bf16(false, a1.v, false, b0.v,
                                                            (short)0, acc[1][0], false, false);
        acc[1][1] = __builtin_amdgcn_wmma_f32_16x16x32_bf16(false, a1.v, false, b1.v,
                                                            (short)0, acc[1][1], false, false);
    }

#pragma unroll
    for (int ni = 0; ni < 2; ++ni) {
        int n    = (nt0 + ni) * 16 + lm;
        float bn = bias[n];
        int sec  = n / CC;                 // 0=q 1=k 2=v
        int f    = n - sec * CC;
        int head = f >> 6, dim = f & 63;
#pragma unroll
        for (int mi = 0; mi < 2; ++mi) {
#pragma unroll
            for (int r = 0; r < 8; ++r) {
                int mg = (mt0 + mi) * 16 + r + 8 * h;      // global token row
                int b0r = mg >> 11, t = mg & (TT - 1);
                float val = acc[mi][ni][r] + bn;
                size_t bh = (size_t)(b0r * HH + head);
                if (sec == 0)      Qb[(bh * TT + t) * HD + dim] = (bf16_t)val;
                else if (sec == 1) Kb[(bh * TT + t) * HD + dim] = (bf16_t)val;
                else               Vt[(bh * HD + dim) * TT + t] = (bf16_t)val;
            }
        }
    }
}

// ---------------------------------------------------------------------------
// 4) Flash attention: one wave per (b*h, 16-query tile). 32-key chunks.
// ---------------------------------------------------------------------------
__global__ void k_flash(const bf16_t* __restrict__ Qb,
                        const bf16_t* __restrict__ Kb,
                        const bf16_t* __restrict__ Vt,
                        bf16_t* __restrict__ yb) {
    __shared__ bf16_t ptile[16 * 32];          // P staging: C/D layout -> A-frag

    int bh = blockIdx.y;
    int q0 = blockIdx.x * 16;
    int L  = threadIdx.x;
    int lm = L & 15, h = L >> 4;

    const bf16_t* Qh = Qb + (size_t)bh * TT * HD;
    const bf16_t* Kh = Kb + (size_t)bh * TT * HD;
    const bf16_t* Vh = Vt + (size_t)bh * HD * TT;

    // Q fragments (HD=64 -> two K=32 steps)
    AF a0, a1;
#pragma unroll
    for (int v = 0; v < 8; ++v) {
        int k0 = apat(v, h);
        a0.u[v] = *(const uint32_t*)(Qh + (size_t)(q0 + lm) * HD + k0);
        a1.u[v] = *(const uint32_t*)(Qh + (size_t)(q0 + lm) * HD + 32 + k0);
    }

    float mrow[8], lrow[8];
    v8f acc[4];
#pragma unroll
    for (int r = 0; r < 8; ++r) { mrow[r] = -1e30f; lrow[r] = 0.0f; }
#pragma unroll
    for (int d = 0; d < 4; ++d) acc[d] = v8f{};

    const float scale = 0.125f;                // 1/sqrt(64)
    int jmax = (q0 + 15) >> 5;

    for (int j = 0; j <= jmax; ++j) {
        // S = Q K^T for 32-key chunk: two 16x16 n-tiles
        v8f s[2];
#pragma unroll
        for (int nt2 = 0; nt2 < 2; ++nt2) {
            AF bk0, bk1;
            int key = j * 32 + nt2 * 16 + lm;
#pragma unroll
            for (int v = 0; v < 8; ++v) {
                int k0 = bpat(v, h);
                bk0.u[v] = *(const uint32_t*)(Kh + (size_t)key * HD + k0);
                bk1.u[v] = *(const uint32_t*)(Kh + (size_t)key * HD + 32 + k0);
            }
            v8f sv = {};
            sv = __builtin_amdgcn_wmma_f32_16x16x32_bf16(false, a0.v, false, bk0.v,
                                                         (short)0, sv, false, false);
            sv = __builtin_amdgcn_wmma_f32_16x16x32_bf16(false, a1.v, false, bk1.v,
                                                         (short)0, sv, false, false);
            s[nt2] = sv;
        }

        // mask + online softmax (row = r + 8*h, col = lane&15 within each n-tile)
        float corr[8];
#pragma unroll
        for (int r = 0; r < 8; ++r) {
            int query = q0 + r + 8 * h;
            int key0  = j * 32 + lm;
            float v0 = (key0      <= query) ? s[0][r] * scale : -1e30f;
            float v1 = (key0 + 16 <= query) ? s[1][r] * scale : -1e30f;
            float mx = fmaxf(v0, v1);
#pragma unroll
            for (int off = 1; off < 16; off <<= 1)
                mx = fmaxf(mx, __shfl_xor(mx, off, 32));
            float mnew = fmaxf(mrow[r], mx);
            float c = __expf(mrow[r] - mnew);
            mrow[r] = mnew;
            float p0 = __expf(v0 - mnew);
            float p1 = __expf(v1 - mnew);
            float rs = p0 + p1;
#pragma unroll
            for (int off = 1; off < 16; off <<= 1)
                rs += __shfl_xor(rs, off, 32);
            lrow[r] = lrow[r] * c + rs;
            corr[r] = c;
            ptile[(r + 8 * h) * 32 + lm]      = (bf16_t)p0;
            ptile[(r + 8 * h) * 32 + 16 + lm] = (bf16_t)p1;
        }
        __syncthreads();   // single-wave WG: just forces DS wait ordering

        // reload P in A-fragment layout
        AF pa;
#pragma unroll
        for (int v = 0; v < 8; ++v)
            pa.u[v] = *(const uint32_t*)&ptile[lm * 32 + apat(v, h)];

        // acc = acc*corr + P @ V  (V pre-transposed: [dim][t], pair-contiguous)
#pragma unroll
        for (int d = 0; d < 4; ++d) {
            v8f t = acc[d];
#pragma unroll
            for (int r = 0; r < 8; ++r) t[r] *= corr[r];
            AF bv;
#pragma unroll
            for (int v = 0; v < 8; ++v) {
                int k0 = bpat(v, h);
                bv.u[v] = *(const uint32_t*)(Vh + (size_t)(d * 16 + lm) * TT + j * 32 + k0);
            }
            acc[d] = __builtin_amdgcn_wmma_f32_16x16x32_bf16(false, pa.v, false, bv.v,
                                                             (short)0, t, false, false);
        }
        __syncthreads();
    }

    // epilogue: normalize, write bf16 y [B*T, C]
    int b0 = bh / HH, head = bh % HH;
#pragma unroll
    for (int d = 0; d < 4; ++d)
#pragma unroll
        for (int r = 0; r < 8; ++r) {
            float y = acc[d][r] / lrow[r];
            size_t row = (size_t)b0 * TT + q0 + r + 8 * h;
            yb[row * CC + head * HD + d * 16 + lm] = (bf16_t)y;
        }
}

// ---------------------------------------------------------------------------
// 5) Proj GEMM, 2x2 register-blocked: y [MM x CC] bf16 x packed Wproj + bias
//    -> fp32 out
// ---------------------------------------------------------------------------
__global__ void k_gemm_proj(const bf16_t* __restrict__ yb,
                            const bf16_t* __restrict__ Wp,
                            const float*  __restrict__ bias,
                            float* __restrict__ out) {
    const int KT = CC / 32;
    int nt0 = blockIdx.x * 2, mt0 = blockIdx.y * 2;
    int L  = threadIdx.x;
    int lm = L & 15, h = L >> 4;

    v8f acc[2][2];
    acc[0][0] = v8f{}; acc[0][1] = v8f{}; acc[1][0] = v8f{}; acc[1][1] = v8f{};

    const bf16_t* ap0 = yb + (size_t)(mt0 * 16 + lm) * CC;
    const bf16_t* ap1 = ap0 + (size_t)16 * CC;
    const bf16_t* bp0 = Wp + (size_t)(nt0 * KT) * 512 + L * 16;
    const bf16_t* bp1 = bp0 + (size_t)KT * 512;

    for (int kt = 0; kt < KT; ++kt) {
        AF a0, a1, b0, b1;
#pragma unroll
        for (int v = 0; v < 8; ++v) {
            int ka = kt * 32 + apat(v, h);
            a0.u[v] = *(const uint32_t*)(ap0 + ka);
            a1.u[v] = *(const uint32_t*)(ap1 + ka);
            b0.u[v] = *(const uint32_t*)(bp0 + kt * 512 + 2 * v);
            b1.u[v] = *(const uint32_t*)(bp1 + kt * 512 + 2 * v);
        }
        acc[0][0] = __builtin_amdgcn_wmma_f32_16x16x32_bf16(false, a0.v, false, b0.v,
                                                            (short)0, acc[0][0], false, false);
        acc[0][1] = __builtin_amdgcn_wmma_f32_16x16x32_bf16(false, a0.v, false, b1.v,
                                                            (short)0, acc[0][1], false, false);
        acc[1][0] = __builtin_amdgcn_wmma_f32_16x16x32_bf16(false, a1.v, false, b0.v,
                                                            (short)0, acc[1][0], false, false);
        acc[1][1] = __builtin_amdgcn_wmma_f32_16x16x32_bf16(false, a1.v, false, b1.v,
                                                            (short)0, acc[1][1], false, false);
    }

#pragma unroll
    for (int ni = 0; ni < 2; ++ni) {
        int n    = (nt0 + ni) * 16 + lm;
        float bn = bias[n];
#pragma unroll
        for (int mi = 0; mi < 2; ++mi) {
#pragma unroll
            for (int r = 0; r < 8; ++r) {
                int mg = (mt0 + mi) * 16 + r + 8 * h;
                out[(size_t)mg * CC + n] = acc[mi][ni][r] + bn;
            }
        }
    }
}

// ---------------------------------------------------------------------------
extern "C" void kernel_launch(void* const* d_in, const int* in_sizes, int n_in,
                              void* d_out, int out_size, void* d_ws, size_t ws_size,
                              hipStream_t stream) {
    const float* x     = (const float*)d_in[0];
    const float* Wqkv  = (const float*)d_in[1];
    const float* bqkv  = (const float*)d_in[2];
    const float* Wproj = (const float*)d_in[3];
    const float* bproj = (const float*)d_in[4];
    float* out = (float*)d_out;

    // workspace carve-up (bf16 buffers)
    const size_t nXB   = (size_t)MM * CC;          // 3,145,728
    const size_t nWQ   = (size_t)CC * NQKV;        // 1,769,472
    const size_t nWP   = (size_t)CC * CC;          // 589,824
    const size_t nHead = (size_t)BB * HH * TT * HD;// 3,145,728 (each of Q,K,Vt)

    char* p = (char*)d_ws;
    auto carve = [&](size_t elems) { bf16_t* q = (bf16_t*)p; p += ((elems * 2 + 255) & ~(size_t)255); return q; };
    bf16_t* xb   = carve(nXB);
    bf16_t* wqp  = carve(nWQ);
    bf16_t* wpp  = carve(nWP);
    bf16_t* Qb   = carve(nHead);
    bf16_t* Kb   = carve(nHead);
    bf16_t* Vt   = carve(nHead);
    bf16_t* ybuf = xb;   // xb is dead after the QKV GEMM; reuse for y

    if ((size_t)((char*)p - (char*)d_ws) > ws_size) return;  // insufficient scratch

    k_cvt_bf16<<<(int)((nXB + 255) / 256), 256, 0, stream>>>(x, xb, (int)nXB);
    k_pack_w<<<(int)((nWQ + 255) / 256), 256, 0, stream>>>(Wqkv, wqp, CC, NQKV);
    k_pack_w<<<(int)((nWP + 255) / 256), 256, 0, stream>>>(Wproj, wpp, CC, CC);

    k_gemm_qkv<<<dim3(NQKV / 32, MM / 32), 32, 0, stream>>>(xb, wqp, bqkv, Qb, Kb, Vt);
    k_flash<<<dim3(TT / 16, BB * HH), 32, 0, stream>>>(Qb, Kb, Vt, ybuf);
    k_gemm_proj<<<dim3(CC / 32, MM / 32), 32, 0, stream>>>(ybuf, wpp, bproj, out);
}
